// Transformer_68556267979287
// MI455X (gfx1250) — matrix operand order repaired and checked by reading, not compiled, not run on hardware
//
#include <hip/hip_runtime.h>

// ---------------------------------------------------------------------------
// Transformer forward for MI455X (gfx1250), full bf16 dataflow.
//   * All GEMMs: v_wmma_f32_16x16x32_bf16 (K=32/instr vs K=4 for f32 WMMA).
//   * Activations and weights live in bf16; f32->bf16 conversion happens only
//     at tensor-production time (GEMM epilogue / tiny repack kernels), never
//     in the GEMM inner loop.
//   * Tile staging uses CDNA5 async copies: global_load_async_to_lds_b128
//     (no VGPR data staging -> no spills; tracked by ASYNCcnt), double
//     buffered so the next tile's fetch overlaps WMMA on the current tile.
// Per-wave tile 32x64: 2 A-frags x 4 B-frags = 8 WMMAs per K-step.
// ---------------------------------------------------------------------------

typedef __bf16 bf16_t;
typedef bf16_t v16bf __attribute__((ext_vector_type(16)));
typedef float  v8f   __attribute__((ext_vector_type(8)));
typedef unsigned short ushort_t;

static constexpr int Dm  = 512;    // model dim
static constexpr int Hh  = 8;      // heads
static constexpr int DKk = 64;     // head dim
static constexpr int Ll  = 6;      // layers
static constexpr int Vv  = 32000;  // vocab
static constexpr int Ss  = 512;    // seq len
static constexpr int Bb  = 4;      // batch
static constexpr int Mtot = Bb * Ss;   // 2048 token rows

#define TM 128
#define TN 64
#define TK 32
#define LROW (TK + 8)   // 40 shorts = 80B row: 16B aligned, 20-bank stride

__device__ __forceinline__ unsigned short f2bf(float f) {
  unsigned int u = __float_as_uint(f);
  u += 0x7fffu + ((u >> 16) & 1u);        // round-to-nearest-even
  return (unsigned short)(u >> 16);
}
__device__ __forceinline__ float bf2f(unsigned short u) {
  return __uint_as_float((unsigned int)u << 16);
}

// CDNA5 async copy: 16B per lane, global -> LDS, tracked by ASYNCcnt.
__device__ __forceinline__ void async_cp16(unsigned lds_addr, const void* gaddr) {
  asm volatile("global_load_async_to_lds_b128 %0, %1, off"
               :: "v"(lds_addr), "v"(gaddr) : "memory");
}
__device__ __forceinline__ void wait_async0() {
  asm volatile("s_wait_asynccnt 0" ::: "memory");
}

union FragAB { v16bf v; unsigned int u[8]; };

// ---------------------------------------------------------------------------
// Batched bf16 WMMA GEMM, B always [N][K] ("nk") bf16:
//   Out[z] = act( alpha * A[z] @ B[z] + bias + residual )
// A: bf16 [M,K] row-major (lda), batch base = sAb*(z/Hdiv)+sAh*(z%Hdiv)
// B: bf16, element (k,n) at n*ldb+k
// Out: bf16 (or f32 if out_f32), optionally transposed store (n-major)
// ---------------------------------------------------------------------------
__global__ __launch_bounds__(128)
void gemm_bf16_wmma(const ushort_t* __restrict__ A, int lda, long sAb, long sAh,
                    const ushort_t* __restrict__ Bw, int ldb, long sBb, long sBh,
                    void* __restrict__ Out, int ldo, long sOb, long sOh,
                    int out_f32, int out_tr,
                    const float* __restrict__ bias,
                    const ushort_t* __restrict__ res,   // bf16, Out layout
                    int M, int N, int K, int Hdiv, float alpha, int relu) {
  __shared__ ushort_t As[2][TM][LROW];
  __shared__ ushort_t Bs[2][TN][LROW];

  const int z  = blockIdx.z;
  const int zb = z / Hdiv, zh = z % Hdiv;
  const ushort_t* Ab = A  + sAb * zb + sAh * zh;
  const ushort_t* Bb_ = Bw + sBb * zb + sBh * zh;
  const long obase = sOb * zb + sOh * zh;

  const int m0 = blockIdx.y * TM;
  const int n0 = blockIdx.x * TN;
  const int tid  = threadIdx.x;
  const int lane = tid & 31;
  const int wave = tid >> 5;       // 0..3 -> 32-row slab
  const int lh   = lane >> 4;      // lane half (0/1)
  const int l16  = lane & 15;

  v8f acc[2][4];
#pragma unroll
  for (int t = 0; t < 2; ++t)
#pragma unroll
    for (int s = 0; s < 4; ++s)
#pragma unroll
      for (int r = 0; r < 8; ++r) acc[t][s][r] = 0.f;

  // --- async staging: each thread issues 6 x 16B global->LDS copies ---
  auto issue = [&](int buf, int k0) {
#pragma unroll
    for (int j = 0; j < 4; ++j) {
      int i = tid + j * 128;               // over TM*TK/8 = 512 groups
      int row = i >> 2, c8 = (i & 3) << 3; // TK/8 = 4 groups/row
      async_cp16((unsigned)(size_t)&As[buf][row][c8],
                 &Ab[(long)(m0 + row) * lda + k0 + c8]);
    }
#pragma unroll
    for (int j = 0; j < 2; ++j) {
      int i = tid + j * 128;               // over TN*TK/8 = 256 groups
      int row = i >> 2, c8 = (i & 3) << 3;
      async_cp16((unsigned)(size_t)&Bs[buf][row][c8],
                 &Bb_[(long)(n0 + row) * ldb + k0 + c8]);
    }
  };
  auto compute = [&](int buf) {
    // A frag: lanes 0-15 -> K 0..7 & 16..23; lanes 16-31 -> 8..15 & 24..31
    FragAB a0, a1;
    const int arow = wave * 32 + l16;
#pragma unroll
    for (int i = 0; i < 8; ++i) {
      int k = ((i & 3) << 1) + (lh << 3) + ((i >> 2) << 4);
      a0.u[i] = *(const unsigned int*)&As[buf][arow][k];
      a1.u[i] = *(const unsigned int*)&As[buf][arow + 16][k];
    }
#pragma unroll
    for (int s = 0; s < 4; ++s) {
      // B frag: lanes 0-15 -> K 0..7 & 16..23; lanes 16-31 -> 8..15 & 24..31
      FragAB bfr;
      const int brow = s * 16 + l16;
#pragma unroll
      for (int i = 0; i < 8; ++i) {
        int k = ((i & 3) << 1) + (lh << 3) + ((i >> 2) << 4);
        bfr.u[i] = *(const unsigned int*)&Bs[buf][brow][k];
      }
      acc[0][s] = __builtin_amdgcn_wmma_f32_16x16x32_bf16(
          false, a0.v, false, bfr.v, (short)0, acc[0][s], false, false);
      acc[1][s] = __builtin_amdgcn_wmma_f32_16x16x32_bf16(
          false, a1.v, false, bfr.v, (short)0, acc[1][s], false, false);
    }
  };

  // --- double-buffered main loop (async fetch overlaps WMMA) ---
  const int ntiles = K / TK;
  issue(0, 0);
  wait_async0();
  __syncthreads();
  for (int kt = 0; kt < ntiles; ++kt) {
    int cur = kt & 1;
    bool more = (kt + 1) < ntiles;
    if (more) issue(1 - cur, (kt + 1) * TK);   // async engine fills back buffer
    compute(cur);
    if (more) {
      wait_async0();      // our wave's copies done
      __syncthreads();    // everyone's copies visible
    }
  }

  // --- epilogue: m = r + 8*(lane>>4), n = lane&15 ---
  float* Of = (float*)Out;
  ushort_t* Ou = (ushort_t*)Out;
#pragma unroll
  for (int t = 0; t < 2; ++t) {
#pragma unroll
    for (int s = 0; s < 4; ++s) {
#pragma unroll
      for (int r = 0; r < 8; ++r) {
        int m = m0 + wave * 32 + t * 16 + r + (lh << 3);
        int n = n0 + s * 16 + l16;
        float v = acc[t][s][r] * alpha;
        if (bias) v += bias[n];
        if (res)  v += bf2f(res[obase + (long)m * ldo + n]);
        if (relu) v = v > 0.f ? v : 0.f;
        if (out_f32)      Of[obase + (long)m * ldo + n] = v;
        else if (out_tr)  Ou[obase + (long)n * ldo + m] = f2bf(v);
        else              Ou[obase + (long)m * ldo + n] = f2bf(v);
      }
    }
  }
}

// ---------------------------------------------------------------------------
// Reductions (wave32 shuffles + LDS across waves)
// ---------------------------------------------------------------------------
__device__ __forceinline__ float blockReduceSum(float v, float* red) {
#pragma unroll
  for (int o = 16; o > 0; o >>= 1) v += __shfl_xor(v, o, 32);
  int lane = threadIdx.x & 31, w = threadIdx.x >> 5;
  __syncthreads();
  if (lane == 0) red[w] = v;
  __syncthreads();
  float tot = 0.f;
  int nw = blockDim.x >> 5;
  for (int i = 0; i < nw; ++i) tot += red[i];
  return tot;
}

__device__ __forceinline__ float blockReduceMax(float v, float* red) {
#pragma unroll
  for (int o = 16; o > 0; o >>= 1) v = fmaxf(v, __shfl_xor(v, o, 32));
  int lane = threadIdx.x & 31, w = threadIdx.x >> 5;
  __syncthreads();
  if (lane == 0) red[w] = v;
  __syncthreads();
  float tot = -3.0e38f;
  int nw = blockDim.x >> 5;
  for (int i = 0; i < nw; ++i) tot = fmaxf(tot, red[i]);
  return tot;
}

// LayerNorm (bf16 in/out, f32 math): one block (256 thr) per row of 512
__global__ __launch_bounds__(256)
void layernorm_k(const ushort_t* __restrict__ x, const float* __restrict__ w,
                 const float* __restrict__ b, ushort_t* __restrict__ out) {
  __shared__ float red[8];
  long row = blockIdx.x;
  const ushort_t* xr = x + row * Dm;
  ushort_t* orow = out + row * Dm;
  int t = threadIdx.x;
  float x0 = bf2f(xr[t]), x1 = bf2f(xr[t + 256]);
  float mean = blockReduceSum(x0 + x1, red) * (1.f / Dm);
  float d0 = x0 - mean, d1 = x1 - mean;
  float var = blockReduceSum(d0 * d0 + d1 * d1, red) * (1.f / Dm);
  float inv = rsqrtf(var + 1e-5f);
  orow[t]       = f2bf(d0 * inv * w[t]       + b[t]);
  orow[t + 256] = f2bf(d1 * inv * w[t + 256] + b[t + 256]);
}

// Masked softmax in place over bf16 rows of length S (256 thr per row)
__global__ __launch_bounds__(256)
void softmax_k(ushort_t* __restrict__ sc, int causal) {
  __shared__ float red[8];
  long row = blockIdx.x;
  ushort_t* r = sc + row * (long)Ss;
  int qpos = (int)(row % Ss);
  int t = threadIdx.x;
  float a0 = bf2f(r[t]), a1 = bf2f(r[t + 256]);
  bool ok0 = (!causal) || (t <= qpos);
  bool ok1 = (!causal) || (t + 256 <= qpos);
  float mx = blockReduceMax(fmaxf(ok0 ? a0 : -3.0e38f, ok1 ? a1 : -3.0e38f), red);
  float e0 = ok0 ? __expf(a0 - mx) : 0.f;
  float e1 = ok1 ? __expf(a1 - mx) : 0.f;
  float inv = 1.f / blockReduceSum(e0 + e1, red);
  r[t]       = f2bf(e0 * inv);
  r[t + 256] = f2bf(e1 * inv);
}

// Embedding gather * sqrt(D) + sinusoidal positional encoding -> bf16
__global__ __launch_bounds__(256)
void embed_k(const int* __restrict__ tok, const float* __restrict__ emb,
             ushort_t* __restrict__ out, float scale) {
  long row = blockIdx.x;             // b*S + s
  int pos = (int)(row % Ss);
  int t = threadIdx.x;
  int token = tok[row];
  const float* e = emb + (long)token * Dm;
  ushort_t* o = out + row * Dm;
  int d0 = 2 * t, d1 = d0 + 1;
  float div = __expf((float)d0 * (-9.210340371976184f / (float)Dm));
  float ang = (float)pos * div;
  o[d0] = f2bf(e[d0] * scale + __sinf(ang));
  o[d1] = f2bf(e[d1] * scale + __cosf(ang));
}

// Per-head QKV weights: [H, D, DK] f32 -> [H*DK][D] bf16 ([N][K])
__global__ __launch_bounds__(256)
void repack_qkv_k(const float* __restrict__ W, ushort_t* __restrict__ WB) {
  int i = blockIdx.x * 256 + threadIdx.x;       // i = n*D + d
  if (i >= Dm * Dm) return;
  int n = i >> 9, d = i & 511;
  int h = n >> 6, kk = n & 63;
  WB[i] = f2bf(W[((long)h * Dm + d) * DKk + kk]);
}

// Generic weight: [K][N] f32 row-major -> [N][K] bf16
__global__ __launch_bounds__(256)
void transpose_cvt_k(const float* __restrict__ W, ushort_t* __restrict__ WB,
                     int K, int N) {
  long i = (long)blockIdx.x * 256 + threadIdx.x;  // i = n*K + k
  if (i >= (long)K * N) return;
  int n = (int)(i / K), k = (int)(i % K);
  WB[i] = f2bf(W[(long)k * N + n]);
}

// ---------------------------------------------------------------------------
// Host orchestration
// ---------------------------------------------------------------------------
extern "C" void kernel_launch(void* const* d_in, const int* in_sizes, int n_in,
                              void* d_out, int out_size, void* d_ws, size_t ws_size,
                              hipStream_t stream) {
  (void)in_sizes; (void)n_in; (void)out_size; (void)ws_size;

  // Flattened input order (setup_inputs insertion order, depth-first):
  // 0 src_emb, 1 tgt_emb,
  // 2..79   enc layers (13 each: Wq Wk Wv Wo bo W1 b1 W2 b2 ln1_w ln1_b ln2_w ln2_b)
  // 80 enc_ln_w, 81 enc_ln_b,
  // 82..201 dec layers (20 each: sa(5) W1 b1 W2 b2 ln1_w ln1_b ln2_w ln2_b ca(5) ln3_w ln3_b)
  // 202 dec_ln_w, 203 dec_ln_b, 204 proj, 205 src, 206 tgt, 207 src_mask, 208 tgt_mask
  auto F = [&](int i) { return (const float*)d_in[i]; };
  const int* src = (const int*)d_in[205];
  const int* tgt = (const int*)d_in[206];

  // ---- workspace carve (bf16 buffers) ----
  const long NXe = (long)Bb * Ss * Dm;          // 1,048,576 elements
  char* p = (char*)d_ws;
  auto carve = [&](long elems) {
    ushort_t* r = (ushort_t*)p;
    p += ((elems * 2 + 255) & ~255L);
    return r;
  };
  ushort_t* X  = carve(NXe);
  ushort_t* XN = carve(NXe);
  ushort_t* Qb = carve(NXe);
  ushort_t* Kb = carve(NXe);
  ushort_t* Vt = carve(NXe);                    // [D][B*S] (transposed V)
  ushort_t* Ob = carve(NXe);
  ushort_t* EN = carve(NXe);
  ushort_t* H1 = carve(4 * NXe);                // FFN hidden
  ushort_t* SC = carve((long)Bb * Hh * Ss * Ss);// scores
  ushort_t* WB = carve((long)Vv * Dm);          // repacked weight (max: proj)

  auto gemm = [&](const ushort_t* A, int lda, long sAb, long sAh,
                  const ushort_t* Bw, int ldb, long sBb, long sBh,
                  void* O, int ldo, long sOb, long sOh, int of32, int otr,
                  const float* bias, const ushort_t* res,
                  int M, int N, int K, int batch, int Hdiv, float alpha, int relu) {
    dim3 g(N / TN, M / TM, batch), blk(128);
    gemm_bf16_wmma<<<g, blk, 0, stream>>>(A, lda, sAb, sAh, Bw, ldb, sBb, sBh,
                                          O, ldo, sOb, sOh, of32, otr,
                                          bias, res, M, N, K, Hdiv, alpha, relu);
  };
  auto ln = [&](const ushort_t* x, const float* w, const float* b, ushort_t* o) {
    layernorm_k<<<Mtot, 256, 0, stream>>>(x, w, b, o);
  };
  auto repT = [&](const float* W, int K, int N) {
    transpose_cvt_k<<<(unsigned)(((long)K * N + 255) / 256), 256, 0, stream>>>(W, WB, K, N);
  };

  const long sBS = (long)Ss * Dm;    // per-batch stride in [B,S,D]
  const long sSS = (long)Ss * Ss;    // per-head stride in scores

  // Multi-head attention: xq/xkv -> outX = residual + MHA(...)
  auto attn = [&](int iWq, int iWk, int iWv, int iWo, int ibo,
                  const ushort_t* xq, const ushort_t* xkv, int causal,
                  const ushort_t* resid, ushort_t* outX) {
    // Q/K: [B,S,D]; V stored transposed -> Vt [D][Mtot]
    repack_qkv_k<<<(Dm * Dm + 255) / 256, 256, 0, stream>>>(F(iWq), WB);
    gemm(xq, Dm, 0, 0, WB, Dm, 0, 0, Qb, Dm, 0, 0, 0, 0, nullptr, nullptr,
         Mtot, Dm, Dm, 1, 1, 1.f, 0);
    repack_qkv_k<<<(Dm * Dm + 255) / 256, 256, 0, stream>>>(F(iWk), WB);
    gemm(xkv, Dm, 0, 0, WB, Dm, 0, 0, Kb, Dm, 0, 0, 0, 0, nullptr, nullptr,
         Mtot, Dm, Dm, 1, 1, 1.f, 0);
    repack_qkv_k<<<(Dm * Dm + 255) / 256, 256, 0, stream>>>(F(iWv), WB);
    gemm(xkv, Dm, 0, 0, WB, Dm, 0, 0, Vt, Mtot, 0, 0, 0, /*tr=*/1, nullptr, nullptr,
         Mtot, Dm, Dm, 1, 1, 1.f, 0);
    // scores[z=(b,h)] = Q_slice @ K_slice^T * DK^-0.5 (K already [n][k])
    gemm(Qb, Dm, sBS, DKk, Kb, Dm, sBS, DKk,
         SC, Ss, (long)Hh * sSS, sSS, 0, 0, nullptr, nullptr,
         Ss, Ss, DKk, Bb * Hh, Hh, 0.125f, 0);
    softmax_k<<<Bb * Hh * Ss, 256, 0, stream>>>(SC, causal);
    // O_slice = P @ V_slice ; Vt gives contiguous-k B tiles
    gemm(SC, Ss, (long)Hh * sSS, sSS, Vt, Mtot, (long)Ss, (long)DKk * Mtot,
         Ob, Dm, sBS, DKk, 0, 0, nullptr, nullptr,
         Ss, DKk, Ss, Bb * Hh, Hh, 1.f, 0);
    // outX = O @ Wo + bo + residual
    repT(F(iWo), Dm, Dm);
    gemm(Ob, Dm, 0, 0, WB, Dm, 0, 0, outX, Dm, 0, 0, 0, 0, F(ibo), resid,
         Mtot, Dm, Dm, 1, 1, 1.f, 0);
  };

  // ---------------- encoder ----------------
  embed_k<<<Mtot, 256, 0, stream>>>(src, F(0), X, 22.627416997969522f);
  for (int l = 0; l < Ll; ++l) {
    int e = 2 + 13 * l;
    ln(X, F(e + 9), F(e + 10), XN);                       // ln1
    attn(e + 0, e + 1, e + 2, e + 3, e + 4, XN, XN, 0, XN, X);
    ln(X, F(e + 11), F(e + 12), XN);                      // ln2
    repT(F(e + 5), Dm, 4 * Dm);
    gemm(XN, Dm, 0, 0, WB, Dm, 0, 0, H1, 4 * Dm, 0, 0, 0, 0,
         F(e + 6), nullptr, Mtot, 4 * Dm, Dm, 1, 1, 1.f, 1);          // FFN1+ReLU
    repT(F(e + 7), 4 * Dm, Dm);
    gemm(H1, 4 * Dm, 0, 0, WB, 4 * Dm, 0, 0, X, Dm, 0, 0, 0, 0,
         F(e + 8), X, Mtot, Dm, 4 * Dm, 1, 1, 1.f, 0);                // FFN2+res
  }
  ln(X, F(80), F(81), EN);                                // enc final LN

  // ---------------- decoder ----------------
  embed_k<<<Mtot, 256, 0, stream>>>(tgt, F(1), X, 22.627416997969522f);
  for (int l = 0; l < Ll; ++l) {
    int d = 82 + 20 * l;
    ln(X, F(d + 9), F(d + 10), XN);                       // ln1
    attn(d + 0, d + 1, d + 2, d + 3, d + 4, XN, XN, /*causal=*/1, XN, X);
    ln(X, F(d + 11), F(d + 12), XN);                      // ln2
    attn(d + 13, d + 14, d + 15, d + 16, d + 17, XN, EN, 0, XN, X);   // cross
    ln(X, F(d + 18), F(d + 19), XN);                      // ln3
    repT(F(d + 5), Dm, 4 * Dm);
    gemm(XN, Dm, 0, 0, WB, Dm, 0, 0, H1, 4 * Dm, 0, 0, 0, 0,
         F(d + 6), nullptr, Mtot, 4 * Dm, Dm, 1, 1, 1.f, 1);
    repT(F(d + 7), 4 * Dm, Dm);
    gemm(H1, 4 * Dm, 0, 0, WB, 4 * Dm, 0, 0, X, Dm, 0, 0, 0, 0,
         F(d + 8), X, Mtot, Dm, 4 * Dm, 1, 1, 1.f, 0);
  }
  ln(X, F(202), F(203), XN);                              // dec final LN

  // logits: [2048,512] @ [512,32000] -> f32 d_out
  repT(F(204), Dm, Vv);
  gemm(XN, Dm, 0, 0, WB, Dm, 0, 0, d_out, Vv, 0, 0, /*f32=*/1, 0, nullptr, nullptr,
       Mtot, Vv, Dm, 1, 1, 1.f, 0);
}
